// NodeLayer_2645699854468
// MI455X (gfx1250) — matrix operand order repaired and checked by reading, not compiled, not run on hardware
//
#include <hip/hip_runtime.h>
#include <hip/hip_fp16.h>

// CDNA5 / gfx1250: wave32, WMMA f32_16x16x32_f16 (f16 A/B, f32 accum).
typedef __attribute__((ext_vector_type(16))) _Float16 v16h;
typedef __attribute__((ext_vector_type(8)))  float    v8f;

#define BN_EPS 1e-5f

union Frag16 { uint4 u4[2]; v16h h; };
union PackH  { _Float16 h[2]; unsigned u; };

// Branch-free ELU: v_exp_f32 (TRANS, co-executes with VALU) + cndmask.
// Absolute error vs expm1f is ~1e-7 near 0 — far below the f16 storage
// quantization of the activations.
static __device__ __forceinline__ float eluf(float x){
  float e = __expf(x) - 1.f;
  return x > 0.f ? x : e;
}
static __device__ __forceinline__ void atomAddF(float* p, float v){ unsafeAtomicAdd(p, v); }

// ---------------------------------------------------------------------------
// Edge histogram: in-degree of `row` (for gathered-x BN stats) and of `col`
// (scatter-mean divisor).
__global__ __launch_bounds__(256)
void hist_kernel(const int* __restrict__ ei, int E, int* cnt_row, int* cnt_col)
{
  int e = blockIdx.x * 256 + threadIdx.x;
  if (e < E){
    atomicAdd(&cnt_row[ei[e]], 1);
    atomicAdd(&cnt_col[ei[E + e]], 1);
  }
}

// ---------------------------------------------------------------------------
// Column sum / sum-of-squares over a [R,64] f32 matrix, optionally weighted
// per-row by an int count (degree-weighted node stats == gathered edge stats).
__global__ __launch_bounds__(256)
void colstats_kernel(const float* __restrict__ X, const int* __restrict__ wcnt,
                     int R, float* __restrict__ sS, float* __restrict__ sQ)
{
  __shared__ float cs[64], cq[64];
  const int t = threadIdx.x;
  if (t < 64){ cs[t] = 0.f; cq[t] = 0.f; }
  __syncthreads();
  const int col = t & 63, rg = t >> 6;
  const int base = blockIdx.x * 128;
  float ls = 0.f, lq = 0.f;
  for (int j = 0; j < 32; ++j){
    int r = base + rg + 4*j;
    if (r < R){
      float w = wcnt ? (float)wcnt[r] : 1.f;
      float v = X[(size_t)r*64 + col];
      ls += w * v; lq += w * v * v;
    }
  }
  atomicAdd(&cs[col], ls); atomicAdd(&cq[col], lq);
  __syncthreads();
  if (t < 64){ atomAddF(&sS[t], cs[t]); atomAddF(&sQ[t], cq[t]); }
}

// ---------------------------------------------------------------------------
// Fold BN into Linear:  W'[k][n] = a[k]*W[k][n] (f16, packed into the CDNA5
// 16-bit B-fragment lane layout), bias'[n] = c[n] + sum_k cc[k]*W[k][n] (f32).
// B layout (32x16 f16): lanes 0-15: N=lane, VGPR v<4 = {K=2v, K=2v+1}, v>=4 = {K=16+2(v-4), +1};
//                       lanes 16-31: N=lane-16, K offset +8.
__global__ __launch_bounds__(256)
void fold_kernel(const float* __restrict__ W, const float* __restrict__ c,
                 const float* __restrict__ g, const float* __restrict__ b,
                 const float* __restrict__ sS, const float* __restrict__ sQ,
                 float cnt, int K, unsigned* __restrict__ Wpk, float* __restrict__ biasF)
{
  __shared__ float la[128], lcc[128];
  const int t = threadIdx.x;
  if (t < K){
    float m = sS[t] / cnt;
    float v = sQ[t] / cnt - m*m;           // biased variance (jnp.var default)
    float a = g[t] * rsqrtf(v + BN_EPS);
    la[t]  = a;
    lcc[t] = b[t] - m*a;
  }
  __syncthreads();
  const int total = K * 32;                // packed dwords: (K/32)*4*32*8
  for (int idx = t; idx < total; idx += 256){
    int v8   = idx & 7;
    int lane = (idx >> 3) & 31;
    int nt   = (idx >> 8) & 3;
    int kt   = idx >> 10;
    int k0   = kt*32 + ((lane >> 4) * 8) + (v8 >> 2) * 16 + 2*(v8 & 3);
    int n    = nt*16 + (lane & 15);
    PackH p;
    p.h[0] = (_Float16)(la[k0]   * W[(size_t)k0*64 + n]);
    p.h[1] = (_Float16)(la[k0+1] * W[(size_t)(k0+1)*64 + n]);
    Wpk[idx] = p.u;
  }
  if (t < 64){
    float acc = c[t];
    for (int k = 0; k < K; ++k) acc += lcc[k] * W[(size_t)k*64 + t];
    biasF[t] = acc;
  }
}

// ---------------------------------------------------------------------------
// Generic fused GEMM: [64 rows x K] f16 (from LDS) @ [K x 64] packed f16
// weights, f32 accumulate via v_wmma_f32_16x16x32_f16.
// MODE 0: +bias, ELU, fused next-layer column stats, store f16.
// MODE 1: +bias, scatter atomicAdd into s[col[e]][n] (f32).
// MODE 2: +bias, store f32 (final output).
template<int K, int MODE>
__global__ __launch_bounds__(128)
void gemm_kernel(const _Float16* __restrict__ Asrc, int R,
                 const unsigned* __restrict__ Wpk, const float* __restrict__ biasF,
                 _Float16* __restrict__ Hout, float* __restrict__ Fout,
                 float* __restrict__ sS, float* __restrict__ sQ,
                 const int* __restrict__ colidx, float* __restrict__ scat)
{
  __shared__ __align__(16) _Float16 ldsA[64 * K];
  __shared__ __align__(16) float    ldsC[64 * 64];
  __shared__ float cs[64], cq[64];
  const int t = threadIdx.x;
  const int rbase = blockIdx.x * 64;

  { // stage 64 x K f16 A-tile, coalesced b128
    const uint4* src = (const uint4*)(Asrc + (size_t)rbase * K);
    uint4* dst = (uint4*)ldsA;
    constexpr int N4  = 64 * K / 8;
    constexpr int RSH = (K == 128) ? 4 : 3;
    for (int i = t; i < N4; i += 128){
      uint4 v = make_uint4(0u,0u,0u,0u);
      if (rbase + (i >> RSH) < R) v = src[i];
      dst[i] = v;
    }
  }
  if (MODE == 0 && t < 64){ cs[t] = 0.f; cq[t] = 0.f; }
  __syncthreads();

  const int lane = t & 31;
  const int nt   = t >> 5;                 // wave -> 16-col tile
  const int n    = nt*16 + (lane & 15);
  const float bias = biasF[n];

  Frag16 bf[K/32];                         // B fragments, L2-hot
  #pragma unroll
  for (int kt = 0; kt < K/32; ++kt){
    const uint4* bp = (const uint4*)(Wpk + (size_t)((kt*4 + nt)*32 + lane) * 8);
    bf[kt].u4[0] = bp[0]; bf[kt].u4[1] = bp[1];
  }

  float ls = 0.f, lq = 0.f;
  const int rA = lane & 15;
  const int kb = (lane >> 4) * 8;          // A frag: halves kb..kb+7, kb+16..kb+23
  #pragma unroll
  for (int rt = 0; rt < 4; ++rt){
    v8f acc = {};
    #pragma unroll
    for (int kt = 0; kt < K/32; ++kt){
      Frag16 af;
      const _Float16* ap = ldsA + (rt*16 + rA)*K + kt*32 + kb;
      af.u4[0] = *(const uint4*)(ap);
      af.u4[1] = *(const uint4*)(ap + 16);
      acc = __builtin_amdgcn_wmma_f32_16x16x32_f16(false, af.h, false, bf[kt].h,
                                                   (short)0, acc, false, false);
    }
    const int m0 = rt*16 + 8*(lane >> 4);  // C layout: M = i + 8*(lane>>4)
    #pragma unroll
    for (int i = 0; i < 8; ++i){
      float v = acc[i] + bias;
      if (MODE == 0){
        v = eluf(v);
        if (rbase + m0 + i < R){ ls += v; lq += v*v; }
        ((_Float16*)ldsC)[(m0+i)*64 + n] = (_Float16)v;
      } else {
        ldsC[(m0+i)*64 + n] = v;
      }
    }
  }
  __syncthreads();

  if (MODE == 0){
    atomicAdd(&cs[n], ls); atomicAdd(&cq[n], lq);
    __syncthreads();
    if (t < 64){ atomAddF(&sS[t], cs[t]); atomAddF(&sQ[t], cq[t]); }
    const unsigned* csrc = (const unsigned*)ldsC;   // 64x64 f16 = 2048 dwords
    unsigned* hdst = (unsigned*)Hout;
    for (int i = t; i < 2048; i += 128){
      int row = i >> 5;
      if (rbase + row < R) hdst[(size_t)(rbase+row)*32 + (i & 31)] = csrc[i];
    }
  } else if (MODE == 1){
    for (int i = t; i < 4096; i += 128){
      int row = i >> 6, cc2 = i & 63;
      int e = rbase + row;
      if (e < R) atomAddF(&scat[(size_t)colidx[e]*64 + cc2], ldsC[i]);
    }
  } else {
    for (int i = t; i < 4096; i += 128){
      int row = i >> 6;
      if (rbase + row < R) Fout[(size_t)(rbase+row)*64 + (i & 63)] = ldsC[i];
    }
  }
}

// ---------------------------------------------------------------------------
// Layer-1 GEMM: A tile gathered on the fly from x[row[e]] || edge_attr[e]
// (f32 -> f16), fused MODE-0 epilogue. K = 128.
__global__ __launch_bounds__(128)
void gemm1_kernel(const float* __restrict__ xg, const float* __restrict__ ea,
                  const int* __restrict__ rowidx, int E,
                  const unsigned* __restrict__ Wpk, const float* __restrict__ biasF,
                  _Float16* __restrict__ Hout, float* __restrict__ sS, float* __restrict__ sQ)
{
  constexpr int K = 128;
  __shared__ __align__(16) _Float16 ldsA[64 * K];
  __shared__ __align__(16) float    ldsC[64 * 64];
  __shared__ float cs[64], cq[64];
  const int t = threadIdx.x;
  const int rbase = blockIdx.x * 64;

  { // gather-stage: 2 threads/row; half 0 = x[row[e]], half 1 = edge_attr[e]
    const int r = t >> 1, half = t & 1;
    const int e = rbase + r;
    const float* src = nullptr;
    if (e < E) src = half ? (ea + (size_t)e*64) : (xg + (size_t)rowidx[e]*64);
    unsigned* dst = (unsigned*)(ldsA + r*128 + half*64);
    #pragma unroll
    for (int j = 0; j < 16; ++j){
      float4 v = src ? ((const float4*)src)[j] : make_float4(0.f,0.f,0.f,0.f);
      PackH p0, p1;
      p0.h[0] = (_Float16)v.x; p0.h[1] = (_Float16)v.y;
      p1.h[0] = (_Float16)v.z; p1.h[1] = (_Float16)v.w;
      dst[2*j] = p0.u; dst[2*j+1] = p1.u;
    }
  }
  if (t < 64){ cs[t] = 0.f; cq[t] = 0.f; }
  __syncthreads();

  const int lane = t & 31;
  const int nt   = t >> 5;
  const int n    = nt*16 + (lane & 15);
  const float bias = biasF[n];

  Frag16 bf[4];
  #pragma unroll
  for (int kt = 0; kt < 4; ++kt){
    const uint4* bp = (const uint4*)(Wpk + (size_t)((kt*4 + nt)*32 + lane) * 8);
    bf[kt].u4[0] = bp[0]; bf[kt].u4[1] = bp[1];
  }

  float ls = 0.f, lq = 0.f;
  const int rA = lane & 15;
  const int kb = (lane >> 4) * 8;
  #pragma unroll
  for (int rt = 0; rt < 4; ++rt){
    v8f acc = {};
    #pragma unroll
    for (int kt = 0; kt < 4; ++kt){
      Frag16 af;
      const _Float16* ap = ldsA + (rt*16 + rA)*K + kt*32 + kb;
      af.u4[0] = *(const uint4*)(ap);
      af.u4[1] = *(const uint4*)(ap + 16);
      acc = __builtin_amdgcn_wmma_f32_16x16x32_f16(false, af.h, false, bf[kt].h,
                                                   (short)0, acc, false, false);
    }
    const int m0 = rt*16 + 8*(lane >> 4);
    #pragma unroll
    for (int i = 0; i < 8; ++i){
      float v = eluf(acc[i] + bias);
      if (rbase + m0 + i < E){ ls += v; lq += v*v; }
      ((_Float16*)ldsC)[(m0+i)*64 + n] = (_Float16)v;
    }
  }
  __syncthreads();
  atomicAdd(&cs[n], ls); atomicAdd(&cq[n], lq);
  __syncthreads();
  if (t < 64){ atomAddF(&sS[t], cs[t]); atomAddF(&sQ[t], cq[t]); }
  const unsigned* csrc = (const unsigned*)ldsC;
  unsigned* hdst = (unsigned*)Hout;
  for (int i = t; i < 2048; i += 128){
    int row = i >> 5;
    if (rbase + row < E) hdst[(size_t)(rbase+row)*32 + (i & 31)] = csrc[i];
  }
}

// ---------------------------------------------------------------------------
// Node-MLP input prep: A4 = f16(concat(x, s/max(cnt,1))) [N,128] + fused
// column stats for BN4.
__global__ __launch_bounds__(256)
void nodeprep_kernel(const float* __restrict__ x, const float* __restrict__ s,
                     const int* __restrict__ cnt_col, int N,
                     _Float16* __restrict__ A4, float* __restrict__ sS, float* __restrict__ sQ)
{
  __shared__ float cs[128], cq[128];
  const int t = threadIdx.x;
  if (t < 128){ cs[t] = 0.f; cq[t] = 0.f; }
  __syncthreads();
  const int col = t & 127, rg = t >> 7;
  const int base = blockIdx.x * 64;
  float ls = 0.f, lq = 0.f;
  for (int j = 0; j < 32; ++j){
    int r = base + rg + 2*j;
    if (r < N){
      float v;
      if (col < 64) v = x[(size_t)r*64 + col];
      else {
        float c = (float)cnt_col[r];
        v = s[(size_t)r*64 + (col - 64)] / fmaxf(c, 1.f);
      }
      A4[(size_t)r*128 + col] = (_Float16)v;
      ls += v; lq += v*v;
    }
  }
  atomicAdd(&cs[col], ls); atomicAdd(&cq[col], lq);
  __syncthreads();
  if (t < 128){ atomAddF(&sS[t], cs[t]); atomAddF(&sQ[t], cq[t]); }
}

// ---------------------------------------------------------------------------
extern "C" void kernel_launch(void* const* d_in, const int* in_sizes, int n_in,
                              void* d_out, int out_size, void* d_ws, size_t ws_size,
                              hipStream_t stream)
{
  const float* x  = (const float*)d_in[0];
  const int*   ei = (const int*)d_in[1];     // [2][E]: row = ei[e], col = ei[E+e]
  const float* ea = (const float*)d_in[2];
  // d_in[3]=u, d_in[4]=batch: unused by the reference.
  const float* g1 = (const float*)d_in[5];  const float* b1 = (const float*)d_in[6];
  const float* W1 = (const float*)d_in[7];  const float* c1 = (const float*)d_in[8];
  const float* g2 = (const float*)d_in[9];  const float* b2 = (const float*)d_in[10];
  const float* W2 = (const float*)d_in[11]; const float* c2 = (const float*)d_in[12];
  const float* g3 = (const float*)d_in[13]; const float* b3 = (const float*)d_in[14];
  const float* W3 = (const float*)d_in[15]; const float* c3 = (const float*)d_in[16];
  const float* g4 = (const float*)d_in[17]; const float* b4 = (const float*)d_in[18];
  const float* W4 = (const float*)d_in[19]; const float* c4 = (const float*)d_in[20];
  const float* g5 = (const float*)d_in[21]; const float* b5 = (const float*)d_in[22];
  const float* W5 = (const float*)d_in[23]; const float* c5 = (const float*)d_in[24];
  const float* g6 = (const float*)d_in[25]; const float* b6 = (const float*)d_in[26];
  const float* W6 = (const float*)d_in[27]; const float* c6 = (const float*)d_in[28];

  const int N = in_sizes[0] / 64;   // 100000
  const int E = in_sizes[2] / 64;   // 1000000

  // ---- workspace layout -------------------------------------------------
  char* ws = (char*)d_ws;
  float*    statsS = (float*)ws;                    // 6 layers x 128
  float*    statsQ = statsS + 6*128;
  float*    biasF  = statsQ + 6*128;                // 6 layers x 64
  unsigned* Wpk    = (unsigned*)(ws + 8192);        // 16384 packed dwords
  size_t off = 131072;
  const size_t npad = (size_t)((N + 63) & ~63) * 4;
  int* cnt_row = (int*)(ws + off); off += npad;
  int* cnt_col = (int*)(ws + off); off += npad;
  off = (off + 255) & ~(size_t)255;
  float* sacc = (float*)(ws + off); off += (size_t)N * 64 * 4;
  const size_t zero_bytes = off;                    // stats + counts + scatter acc
  off = (off + 255) & ~(size_t)255;
  const size_t regA = (size_t)E*64*2 > (size_t)N*128*2 ? (size_t)E*64*2 : (size_t)N*128*2;
  const size_t regB = (size_t)E*64*2 > (size_t)N*64*2  ? (size_t)E*64*2 : (size_t)N*64*2;
  _Float16* H1 = (_Float16*)(ws + off);             // also A4 (disjoint in time)
  _Float16* A4 = H1;
  off += (regA + 255) & ~(size_t)255;
  _Float16* H2 = (_Float16*)(ws + off);             // also H5 (disjoint in time)
  _Float16* H5 = H2;
  off += (regB + 255) & ~(size_t)255;
  _Float16* H6 = (_Float16*)(ws + off);
  off += (size_t)N * 64 * 2;
  if (off > ws_size) return;                        // workspace too small: bail

  hipMemsetAsync(d_ws, 0, zero_bytes, stream);

  const int gE = (E + 63) / 64, gN = (N + 63) / 64;

  // ---- edge MLP ---------------------------------------------------------
  hist_kernel<<<(E + 255) / 256, 256, 0, stream>>>(ei, E, cnt_row, cnt_col);
  colstats_kernel<<<(N + 127) / 128, 256, 0, stream>>>(x,  cnt_row, N, statsS + 0,  statsQ + 0);
  colstats_kernel<<<(E + 127) / 128, 256, 0, stream>>>(ea, nullptr, E, statsS + 64, statsQ + 64);
  fold_kernel<<<1, 256, 0, stream>>>(W1, c1, g1, b1, statsS + 0, statsQ + 0, (float)E, 128, Wpk + 0, biasF + 0);
  gemm1_kernel<<<gE, 128, 0, stream>>>(x, ea, ei, E, Wpk + 0, biasF + 0, H1, statsS + 128, statsQ + 128);
  fold_kernel<<<1, 256, 0, stream>>>(W2, c2, g2, b2, statsS + 128, statsQ + 128, (float)E, 64, Wpk + 4096, biasF + 64);
  gemm_kernel<64,0><<<gE, 128, 0, stream>>>(H1, E, Wpk + 4096, biasF + 64, H2, nullptr,
                                            statsS + 256, statsQ + 256, nullptr, nullptr);
  fold_kernel<<<1, 256, 0, stream>>>(W3, c3, g3, b3, statsS + 256, statsQ + 256, (float)E, 64, Wpk + 6144, biasF + 128);
  gemm_kernel<64,1><<<gE, 128, 0, stream>>>(H2, E, Wpk + 6144, biasF + 128, nullptr, nullptr,
                                            nullptr, nullptr, ei + E, sacc);

  // ---- node MLP ---------------------------------------------------------
  nodeprep_kernel<<<gN, 256, 0, stream>>>(x, sacc, cnt_col, N, A4, statsS + 384, statsQ + 384);
  fold_kernel<<<1, 256, 0, stream>>>(W4, c4, g4, b4, statsS + 384, statsQ + 384, (float)N, 128, Wpk + 8192, biasF + 192);
  gemm_kernel<128,0><<<gN, 128, 0, stream>>>(A4, N, Wpk + 8192, biasF + 192, H5, nullptr,
                                             statsS + 512, statsQ + 512, nullptr, nullptr);
  fold_kernel<<<1, 256, 0, stream>>>(W5, c5, g5, b5, statsS + 512, statsQ + 512, (float)N, 64, Wpk + 12288, biasF + 256);
  gemm_kernel<64,0><<<gN, 128, 0, stream>>>(H5, N, Wpk + 12288, biasF + 256, H6, nullptr,
                                            statsS + 640, statsQ + 640, nullptr, nullptr);
  fold_kernel<<<1, 256, 0, stream>>>(W6, c6, g6, b6, statsS + 640, statsQ + 640, (float)N, 64, Wpk + 14336, biasF + 320);
  gemm_kernel<64,2><<<gN, 128, 0, stream>>>(H6, N, Wpk + 14336, biasF + 320, nullptr, (float*)d_out,
                                            nullptr, nullptr, nullptr, nullptr);
  (void)n_in; (void)out_size;
}
